// RoleModulator_31447750542012
// MI455X (gfx1250) — matrix operand order repaired and checked by reading, not compiled, not run on hardware
//
#include <hip/hip_runtime.h>

typedef __attribute__((ext_vector_type(16))) _Float16 v16h;
typedef __attribute__((ext_vector_type(8)))  _Float16 v8h;
typedef __attribute__((ext_vector_type(4)))  _Float16 v4h;
typedef __attribute__((ext_vector_type(8)))  float    v8f;

namespace {
constexpr int kB = 8;
constexpr int kS = 2048;
constexpr int kD = 2048;
constexpr int kR = 4;
constexpr int TM = 128;   // output rows per block
constexpr int TN = 128;   // output cols per block
constexpr int TK = 32;    // K per step (one f16 WMMA)
constexpr int STR = 40;   // padded f16 row stride: 80B -> 16B aligned, 16 distinct LDS banks
constexpr int KSTEPS = kD / TK;  // 64
constexpr int PF_DIST = 2;       // prefetch distance in K-steps
}

__device__ __forceinline__ float fast_tanh(float x) {
  float ax = __builtin_fabsf(x);
#if __has_builtin(__builtin_amdgcn_exp2f) && __has_builtin(__builtin_amdgcn_rcpf)
  float e = __builtin_amdgcn_exp2f(ax * -2.8853900817779268f);  // exp(-2*ax)
  float r = (1.0f - e) * __builtin_amdgcn_rcpf(1.0f + e);
#else
  float e = __expf(-2.0f * ax);
  float r = (1.0f - e) / (1.0f + e);
#endif
  return x < 0.0f ? -r : r;
}

__device__ __forceinline__ void store_nt(float* p, float v) {
#if __has_builtin(__builtin_nontemporal_store)
  __builtin_nontemporal_store(v, p);
#else
  *p = v;
#endif
}

// fp32 -> f16 hi/lo split: x ~= (float)h + (float)l, |l| <= 2^-11 |x|
__device__ __forceinline__ void split_f32(float x, _Float16& h, _Float16& l) {
  h = (_Float16)x;
  l = (_Float16)(x - (float)h);
}

__device__ __forceinline__ float f4c(float4 v, int c) {
  return c == 0 ? v.x : (c == 1 ? v.y : (c == 2 ? v.z : v.w));
}

__global__ __launch_bounds__(256, 1)
void role_gemm_tanh_wmma(const float* __restrict__ hidden,
                         const float* __restrict__ onehot,
                         const float* __restrict__ Wall,
                         const float* __restrict__ ball,
                         const float* __restrict__ scl,
                         float* __restrict__ out)
{
  // Double-buffered, pre-split f16 tiles. B is staged transposed (BT[n][k]) so
  // both A and B fragments are contiguous-K 16B LDS reads.
  __shared__ alignas(16) _Float16 sAhi[2][TM * STR];
  __shared__ alignas(16) _Float16 sAlo[2][TM * STR];
  __shared__ alignas(16) _Float16 sBhi[2][TN * STR];
  __shared__ alignas(16) _Float16 sBlo[2][TN * STR];

  const int tid = threadIdx.x;
  const int n0 = blockIdx.x * TN;
  const int s0 = blockIdx.y * TM;
  const int z  = blockIdx.z;

  // one-hot -> role index (R=4, uniform per block)
  int role = 0;
#pragma unroll
  for (int j = 1; j < kR; ++j) role = (onehot[z * kR + j] > 0.5f) ? j : role;

  const float* Ag = hidden + (size_t)z * kS * kD;
  const float* Wg = Wall + (size_t)role * kD * kD;

  // A staging: 2 threads per row, each covers 16 contiguous K (4 float4 loads)
  const int arow = tid >> 1;
  const int akh  = (tid & 1) * 16;
  const float* aP = Ag + (size_t)(s0 + arow) * kD + akh;

  // B staging: thread t owns a 4x4 micro-tile (k=4*kq.., n=4*nq..) and
  // transposes it into BT so stores are 8B-contiguous along K.
  const int kq = tid >> 5;       // 0..7
  const int nq = tid & 31;       // 0..31
  const float* bP = Wg + (size_t)(4 * kq) * kD + (n0 + 4 * nq);

  float4 aR[4], bR[4];

  auto gload = [&](int kt) {
    const float* ap = aP + kt * TK;
#pragma unroll
    for (int i = 0; i < 4; ++i) aR[i] = *(const float4*)(ap + 4 * i);
    const float* bp = bP + (size_t)kt * TK * kD;
#pragma unroll
    for (int j = 0; j < 4; ++j) bR[j] = *(const float4*)(bp + (size_t)j * kD);
  };

  auto stage = [&](int buf) {
    // A: 16 contiguous K values -> two v8h stores per precision
    v8h ah[2], al[2];
#pragma unroll
    for (int i = 0; i < 4; ++i) {
#pragma unroll
      for (int c = 0; c < 4; ++c) {
        _Float16 hh, ll;
        split_f32(f4c(aR[i], c), hh, ll);
        const int e = i * 4 + c;
        ah[e >> 3][e & 7] = hh;
        al[e >> 3][e & 7] = ll;
      }
    }
    {
      v8h* dh = (v8h*)&sAhi[buf][arow * STR + akh];
      v8h* dl = (v8h*)&sAlo[buf][arow * STR + akh];
      dh[0] = ah[0]; dh[1] = ah[1];
      dl[0] = al[0]; dl[1] = al[1];
    }
    // B: transpose 4x4 micro-tile; column c of the float4s -> 4 contiguous K
#pragma unroll
    for (int c = 0; c < 4; ++c) {
      v4h hh, ll;
#pragma unroll
      for (int j = 0; j < 4; ++j) {
        _Float16 a, b;
        split_f32(f4c(bR[j], c), a, b);
        hh[j] = a; ll[j] = b;
      }
      const int row = 4 * nq + c;
      *(v4h*)&sBhi[buf][row * STR + 4 * kq] = hh;
      *(v4h*)&sBlo[buf][row * STR + 4 * kq] = ll;
    }
  };

  // wave -> 32x64 subtile of the 128x128 block tile
  const int lane = tid & 31;
  const int wid  = tid >> 5;
  const int wy   = wid >> 1;   // 0..3 : row group of 32
  const int wx   = wid & 1;    // 0..1 : col group of 64
  const int lm   = lane & 15;
  const int lh   = lane >> 4;

  v8f acc[2][4];
#pragma unroll
  for (int mt = 0; mt < 2; ++mt)
#pragma unroll
    for (int nt = 0; nt < 4; ++nt)
#pragma unroll
      for (int i = 0; i < 8; ++i) acc[mt][nt][i] = 0.0f;

  gload(0);
  stage(0);
  __syncthreads();

  for (int kt = 0; kt < KSTEPS; ++kt) {
    const int cur = kt & 1;
    if (kt + 1 < KSTEPS) gload(kt + 1);

    // Prefetch the streaming A tile a couple of K-steps ahead
    // (lowers to global_prefetch_b8 on gfx1250).
    if (kt + PF_DIST < KSTEPS) {
      __builtin_prefetch(aP + (kt + PF_DIST) * TK, 0, 3);
    }

    // B fragments (same chunked-K layout as A: lanes0-15 K0-7 & K16-23, lanes16-31 K8-15 & K24-31)
    v16h bh[4], bl[4];
#pragma unroll
    for (int nt = 0; nt < 4; ++nt) {
      const _Float16* ph = &sBhi[cur][(wx * 64 + nt * 16 + lm) * STR + 8 * lh];
      const _Float16* pl = &sBlo[cur][(wx * 64 + nt * 16 + lm) * STR + 8 * lh];
      v8h h0 = *(const v8h*)(ph);
      v8h h1 = *(const v8h*)(ph + 16);
      v8h l0 = *(const v8h*)(pl);
      v8h l1 = *(const v8h*)(pl + 16);
#pragma unroll
      for (int i = 0; i < 8; ++i) {
        bh[nt][i] = h0[i]; bh[nt][8 + i] = h1[i];
        bl[nt][i] = l0[i]; bl[nt][8 + i] = l1[i];
      }
    }

#pragma unroll
    for (int mt = 0; mt < 2; ++mt) {
      // A fragment (ISA 16-bit A 16x32 chunked-K layout:
      // lanes0-15: K0-7 & K16-23, lanes16-31: K8-15 & K24-31)
      const _Float16* ph = &sAhi[cur][(wy * 32 + mt * 16 + lm) * STR + 8 * lh];
      const _Float16* pl = &sAlo[cur][(wy * 32 + mt * 16 + lm) * STR + 8 * lh];
      v8h h0 = *(const v8h*)(ph);
      v8h h1 = *(const v8h*)(ph + 16);
      v8h l0 = *(const v8h*)(pl);
      v8h l1 = *(const v8h*)(pl + 16);
      v16h ah, al;
#pragma unroll
      for (int i = 0; i < 8; ++i) {
        ah[i] = h0[i]; ah[8 + i] = h1[i];
        al[i] = l0[i]; al[8 + i] = l1[i];
      }

#pragma unroll
      for (int nt = 0; nt < 4; ++nt) {
        acc[mt][nt] = __builtin_amdgcn_wmma_f32_16x16x32_f16(
            false, ah, false, bh[nt], (short)0, acc[mt][nt], false, false);
        acc[mt][nt] = __builtin_amdgcn_wmma_f32_16x16x32_f16(
            false, ah, false, bl[nt], (short)0, acc[mt][nt], false, false);
        acc[mt][nt] = __builtin_amdgcn_wmma_f32_16x16x32_f16(
            false, al, false, bh[nt], (short)0, acc[mt][nt], false, false);
      }
    }

    if (kt + 1 < KSTEPS) stage(cur ^ 1);
    __syncthreads();
  }

  // Epilogue: bias + tanh + scale. C/D layout: VGPR r -> M=r (lanes 0-15) or
  // M=r+8 (lanes 16-31), N = lane&15. Output is write-once: store non-temporal
  // so it doesn't evict the reused W tiles from L2.
  const float s = scl[role];
  const float* bv = ball + role * kD;
  float* outB = out + (size_t)z * kS * kD;
#pragma unroll
  for (int mt = 0; mt < 2; ++mt) {
#pragma unroll
    for (int nt = 0; nt < 4; ++nt) {
      const int col = n0 + wx * 64 + nt * 16 + lm;
      const float bias = bv[col];
      const int rbase = s0 + wy * 32 + mt * 16 + 8 * lh;
#pragma unroll
      for (int rr = 0; rr < 8; ++rr) {
        float v = acc[mt][nt][rr] + bias;
        store_nt(&outB[(size_t)(rbase + rr) * kD + col], s * fast_tanh(v));
      }
    }
  }
}

extern "C" void kernel_launch(void* const* d_in, const int* in_sizes, int n_in,
                              void* d_out, int out_size, void* d_ws, size_t ws_size,
                              hipStream_t stream) {
  const float* hidden = (const float*)d_in[0];
  const float* onehot = (const float*)d_in[1];
  const float* W      = (const float*)d_in[2];
  const float* b      = (const float*)d_in[3];
  const float* s      = (const float*)d_in[4];
  float* out = (float*)d_out;

  dim3 grid(kD / TN, kS / TM, kB);   // 16 x 16 x 8 = 2048 blocks
  role_gemm_tanh_wmma<<<grid, dim3(256), 0, stream>>>(hidden, onehot, W, b, s, out);
}
